// PointGNN_22625887715635
// MI455X (gfx1250) — matrix operand order repaired and checked by reading, not compiled, not run on hardware
//
#include <hip/hip_runtime.h>
#include <hip/hip_bf16.h>

typedef __attribute__((ext_vector_type(16))) _Float16 v16h;
typedef __attribute__((ext_vector_type(8)))  _Float16 v8h;
typedef __attribute__((ext_vector_type(8)))  float    v8f;

#define NNODE 100000
#define NEDGE 1600000
#define NGRAPH 64
#define DD 64

// ---------------------------------------------------------------------------
// utility fill
// ---------------------------------------------------------------------------
__global__ void k_fill_u32(unsigned* __restrict__ p, unsigned val, long n) {
    long i = (long)blockIdx.x * blockDim.x + threadIdx.x;
    if (i < n) p[i] = val;
}

// ---------------------------------------------------------------------------
// weight repack: f32 row-major W -> f16 WMMA B-fragment layout
// B frag, contiguous k order used by this code (A matches): lane l holds column N = nbase + (l%16),
// K = kbase + (l>=16?16:0) + 2v (+1 in high half of dword v), v = 0..7
// GEMM1: K<64 -> fw1 row 3+K (x part); 64<=K<67 -> fw1 row K-64 (rel); else 0
// ---------------------------------------------------------------------------
__device__ __forceinline__ unsigned pack2h(float a, float b) {
    union { _Float16 h[2]; unsigned u; } uu;
    uu.h[0] = (_Float16)a; uu.h[1] = (_Float16)b;
    return uu.u;
}
__device__ __forceinline__ float w1val(const float* __restrict__ fw1, int K, int col) {
    if (K < 64) return fw1[(3 + K) * 64 + col];
    if (K < 67) return fw1[(K - 64) * 64 + col];
    return 0.f;
}
__global__ void k_prepw(const float* __restrict__ fw1, const float* __restrict__ fw2,
                        unsigned* __restrict__ wp1, unsigned* __restrict__ wp2) {
    int id = blockIdx.x * blockDim.x + threadIdx.x;
    if (id < 12 * 32 * 8) {                       // GEMM1: 12 frags (4 ntile x 3 kchunk)
        int v = id & 7, lanei = (id >> 3) & 31, f = id >> 8;
        int t = f / 3, c = f % 3;
        int col = t * 16 + (lanei & 15);
        int K0  = c * 32 + ((lanei >= 16) ? 16 : 0) + 2 * v;
        wp1[id] = pack2h(w1val(fw1, K0, col), w1val(fw1, K0 + 1, col));
    } else if (id < 12 * 32 * 8 + 8 * 32 * 8) {   // GEMM2: 8 frags (4 ntile x 2 kchunk)
        int id2 = id - 12 * 32 * 8;
        int v = id2 & 7, lanei = (id2 >> 3) & 31, f = id2 >> 8;
        int t = f / 2, c = f % 2;
        int col = t * 16 + (lanei & 15);
        int K0  = c * 32 + ((lanei >= 16) ? 16 : 0) + 2 * v;
        wp2[id2] = pack2h(fw2[K0 * 64 + col], fw2[(K0 + 1) * 64 + col]);
    }
}

// ---------------------------------------------------------------------------
// feature embed: x = h @ fe_w + fe_b ; also f16 copy
// ---------------------------------------------------------------------------
__global__ void k_embed(const float* __restrict__ h, const float* __restrict__ fw,
                        const float* __restrict__ fb, float* __restrict__ x,
                        _Float16* __restrict__ xh, int n) {
    int i = blockIdx.x * blockDim.x + threadIdx.x;
    if (i >= n) return;
    float h0 = h[i*4+0], h1 = h[i*4+1], h2 = h[i*4+2], h3 = h[i*4+3];
    for (int d = 0; d < 64; ++d) {
        float v = fb[d] + h0*fw[d] + h1*fw[64+d] + h2*fw[128+d] + h3*fw[192+d];
        x[(size_t)i*64 + d]  = v;
        xh[(size_t)i*64 + d] = (_Float16)v;
    }
}

// ---------------------------------------------------------------------------
// delta MLP per node, store q = delta - pos  (rel = pos[src] + q[dst])
// ---------------------------------------------------------------------------
__global__ void k_delta(const float* __restrict__ x, const float* __restrict__ pos,
                        const float* __restrict__ w1, const float* __restrict__ b1,
                        const float* __restrict__ w2, const float* __restrict__ b2,
                        float* __restrict__ q, int n) {
    int i = blockIdx.x * blockDim.x + threadIdx.x;
    if (i >= n) return;
    float xi[64];
    for (int k = 0; k < 64; ++k) xi[k] = x[(size_t)i*64 + k];
    float d0 = b2[0], d1 = b2[1], d2 = b2[2];
    for (int j = 0; j < 64; ++j) {
        float t = b1[j];
        for (int k = 0; k < 64; ++k) t += xi[k] * w1[k*64 + j];
        t = fmaxf(t, 0.f);
        d0 += t * w2[j*3+0]; d1 += t * w2[j*3+1]; d2 += t * w2[j*3+2];
    }
    q[i*3+0] = d0 - pos[i*3+0];
    q[i*3+1] = d1 - pos[i*3+1];
    q[i*3+2] = d2 - pos[i*3+2];
}

// ---------------------------------------------------------------------------
// sign-trick float atomic max (lowers to native i32 max / u32 min atomics)
// ---------------------------------------------------------------------------
__device__ __forceinline__ void atomicMaxF(float* addr, float v) {
    if (!__builtin_signbitf(v)) atomicMax((int*)addr, __float_as_int(v));
    else                        atomicMin((unsigned int*)addr, __float_as_uint(v));
}

// ---------------------------------------------------------------------------
// fused edge kernel: msg = relu(A@W1+b1)@W2+b2 via WMMA, atomic-max scatter
// A row (K dim): [ x[src] (0..63) | rel (64..66) | 0 pad (67..95) ]
// ---------------------------------------------------------------------------
#define NWAVE 8
__global__ __launch_bounds__(256)
void k_edge(const int* __restrict__ src, const int* __restrict__ dst,
            const float* __restrict__ pos, const float* __restrict__ q,
            const _Float16* __restrict__ xh,
            const unsigned* __restrict__ wp1, const unsigned* __restrict__ wp2,
            const float* __restrict__ fb1, const float* __restrict__ fb2,
            float* __restrict__ agg, int ntiles) {
    __shared__ _Float16 sA1[NWAVE][16 * 104];   // padded stride: bank-conflict free
    __shared__ _Float16 sA2[NWAVE][16 * 72];
    __shared__ int      sDst[NWAVE][16];

    const int lane  = threadIdx.x & 31;
    const int w     = threadIdx.x >> 5;
    const int gwave = blockIdx.x * NWAVE + w;
    const int nwav  = gridDim.x * NWAVE;
    const int half  = (lane >> 4) & 1;
    const int col   = lane & 15;

    // B fragments resident in VGPRs for the whole persistent loop
    v16h B1[12], B2[8];
    for (int f = 0; f < 12; ++f) B1[f] = *(const v16h*)(wp1 + (f*32 + lane)*8);
    for (int f = 0; f < 8;  ++f) B2[f] = *(const v16h*)(wp2 + (f*32 + lane)*8);
    float b1v[4], b2v[4];
    for (int t = 0; t < 4; ++t) { b1v[t] = fb1[t*16 + col]; b2v[t] = fb2[t*16 + col]; }

    // zero the static pad region K = 64..103 once (rel rewrites 64..66 each tile)
    if (lane < 16) {
        float4* rowp = (float4*)(&sA1[w][lane*104 + 64]);
        for (int i = 0; i < 5; ++i) rowp[i] = make_float4(0.f, 0.f, 0.f, 0.f);
    }
    asm volatile("s_wait_dscnt 0" ::: "memory");

    for (int tile = gwave; tile < ntiles; tile += nwav) {
        const int ebase = tile * 16;

        // --- stage rel + dst (one edge per lane 0..15) ---
        if (lane < 16) {
            int e = ebase + lane;
            int s = src[e], d0 = dst[e];
            sDst[w][lane] = d0;
            _Float16* row = &sA1[w][lane*104 + 64];
            row[0] = (_Float16)(pos[s*3+0] + q[d0*3+0]);
            row[1] = (_Float16)(pos[s*3+1] + q[d0*3+1]);
            row[2] = (_Float16)(pos[s*3+2] + q[d0*3+2]);
        }
        // --- stage x[src]: 2 lanes per edge row, 64B each ---
        {
            int r = lane >> 1, hh = lane & 1;
            int s = src[ebase + r];
            const float4* gsrc = (const float4*)(xh + (size_t)s*64) + hh*4;
            float4* ldst = (float4*)(&sA1[w][r*104]) + hh*4;
            for (int i = 0; i < 4; ++i) ldst[i] = gsrc[i];
        }
        asm volatile("s_wait_dscnt 0" ::: "memory");

        // --- GEMM1: [16x96] @ [96x64], f16 -> f32 ---
        v8f C1[4];
        #pragma unroll
        for (int t = 0; t < 4; ++t) {
            v8f c; 
            #pragma unroll
            for (int j = 0; j < 8; ++j) c[j] = b1v[t];
            C1[t] = c;
        }
        const char* abase = (const char*)&sA1[w][0] + col*208 + half*32;
        #pragma unroll
        for (int c = 0; c < 3; ++c) {
            const char* p = abase + c*64;
            v8h lo = *(const v8h*)(p);
            v8h hi = *(const v8h*)(p + 16);
            v16h a = __builtin_shufflevector(lo, hi, 0,1,2,3,4,5,6,7,8,9,10,11,12,13,14,15);
            #pragma unroll
            for (int t = 0; t < 4; ++t)
                C1[t] = __builtin_amdgcn_wmma_f32_16x16x32_f16(
                            false, a, false, B1[t*3 + c], (short)0, C1[t], false, false);
        }

        // --- ReLU + transpose through LDS (C layout -> A layout) ---
        {
            int M0 = half ? 8 : 0;
            #pragma unroll
            for (int t = 0; t < 4; ++t)
                #pragma unroll
                for (int j = 0; j < 8; ++j) {
                    float v = C1[t][j];
                    v = v > 0.f ? v : 0.f;
                    sA2[w][(M0 + j)*72 + t*16 + col] = (_Float16)v;
                }
        }
        asm volatile("s_wait_dscnt 0" ::: "memory");

        // --- GEMM2: [16x64] @ [64x64] ---
        v8f C2[4];
        #pragma unroll
        for (int t = 0; t < 4; ++t) {
            v8f c;
            #pragma unroll
            for (int j = 0; j < 8; ++j) c[j] = b2v[t];
            C2[t] = c;
        }
        const char* a2base = (const char*)&sA2[w][0] + col*144 + half*32;
        #pragma unroll
        for (int c = 0; c < 2; ++c) {
            const char* p = a2base + c*64;
            v8h lo = *(const v8h*)(p);
            v8h hi = *(const v8h*)(p + 16);
            v16h a = __builtin_shufflevector(lo, hi, 0,1,2,3,4,5,6,7,8,9,10,11,12,13,14,15);
            #pragma unroll
            for (int t = 0; t < 4; ++t)
                C2[t] = __builtin_amdgcn_wmma_f32_16x16x32_f16(
                            false, a, false, B2[t*2 + c], (short)0, C2[t], false, false);
        }

        // --- segment-max scatter ---
        {
            int M0 = half ? 8 : 0;
            #pragma unroll
            for (int t = 0; t < 4; ++t)
                #pragma unroll
                for (int j = 0; j < 8; ++j) {
                    int d0 = sDst[w][M0 + j];
                    atomicMaxF(agg + (size_t)d0*64 + t*16 + col, C2[t][j]);
                }
        }
    }
}

// ---------------------------------------------------------------------------
// g = relu(agg @ gw1 + gb1); accumulate batch-norm sum / sumsq
// ---------------------------------------------------------------------------
__global__ __launch_bounds__(256)
void k_g1(const float* __restrict__ agg, const float* __restrict__ w1,
          const float* __restrict__ b1, float* __restrict__ tbuf,
          float* __restrict__ stats, int n) {
    __shared__ float ssum[128];
    for (int k = threadIdx.x; k < 128; k += 256) ssum[k] = 0.f;
    __syncthreads();
    int i = blockIdx.x * 256 + threadIdx.x;
    if (i < n) {
        float ai[64];
        for (int k = 0; k < 64; ++k) {
            float v = agg[(size_t)i*64 + k];
            if (__float_as_uint(v) == 0xFF800000u) v = 0.f;   // empty segment -> 0
            ai[k] = v;
        }
        for (int j = 0; j < 64; ++j) {
            float t = b1[j];
            for (int k = 0; k < 64; ++k) t += ai[k] * w1[k*64 + j];
            t = fmaxf(t, 0.f);
            tbuf[(size_t)i*64 + j] = t;
            atomicAdd(&ssum[j], t);
            atomicAdd(&ssum[64 + j], t * t);
        }
    }
    __syncthreads();
    for (int k = threadIdx.x; k < 128; k += 256) atomicAdd(&stats[k], ssum[k]);
}

// ---------------------------------------------------------------------------
// BN apply + second projection + residual; refresh x / xh
// ---------------------------------------------------------------------------
__global__ void k_bn(const float* __restrict__ tbuf, const float* __restrict__ stats,
                     const float* __restrict__ gm, const float* __restrict__ bt,
                     const float* __restrict__ w2, const float* __restrict__ b2,
                     float* __restrict__ x, _Float16* __restrict__ xh, int n) {
    int i = blockIdx.x * blockDim.x + threadIdx.x;
    if (i >= n) return;
    const float invN = 1.f / (float)NNODE;
    float g[64];
    for (int d = 0; d < 64; ++d) {
        float m  = stats[d] * invN;
        float v  = stats[64 + d] * invN - m * m;
        float rs = rsqrtf(v + 1e-5f);
        g[d] = (tbuf[(size_t)i*64 + d] - m) * rs * gm[d] + bt[d];
    }
    for (int j = 0; j < 64; ++j) {
        float o = b2[j];
        for (int d = 0; d < 64; ++d) o += g[d] * w2[d*64 + j];
        float nv = x[(size_t)i*64 + j] + o;
        x[(size_t)i*64 + j]  = nv;
        xh[(size_t)i*64 + j] = (_Float16)nv;
    }
}

// ---------------------------------------------------------------------------
// global mean pool (sum + count)
// ---------------------------------------------------------------------------
__global__ void k_pool(const float* __restrict__ x, const int* __restrict__ batch,
                       float* __restrict__ pool, float* __restrict__ cnt, int n) {
    int i = blockIdx.x * blockDim.x + threadIdx.x;
    if (i >= n) return;
    int b = batch[i];
    for (int d = 0; d < 64; ++d) atomicAdd(&pool[b*64 + d], x[(size_t)i*64 + d]);
    atomicAdd(&cnt[b], 1.f);
}

// ---------------------------------------------------------------------------
// readout MLP: [G,64] -> [G,16]
// ---------------------------------------------------------------------------
__global__ void k_readout(const float* __restrict__ pool, const float* __restrict__ cnt,
                          const float* __restrict__ w1, const float* __restrict__ b1,
                          const float* __restrict__ w2, const float* __restrict__ b2,
                          float* __restrict__ out) {
    int g = threadIdx.x;
    if (g >= NGRAPH) return;
    float c = fmaxf(cnt[g], 1.f);
    float p[64], t[64];
    for (int d = 0; d < 64; ++d) p[d] = pool[g*64 + d] / c;
    for (int j = 0; j < 64; ++j) {
        float a = b1[j];
        for (int d = 0; d < 64; ++d) a += p[d] * w1[d*64 + j];
        t[j] = fmaxf(a, 0.f);
    }
    for (int o = 0; o < 16; ++o) {
        float a = b2[o];
        for (int j = 0; j < 64; ++j) a += t[j] * w2[j*16 + o];
        out[g*16 + o] = a;
    }
}

// ---------------------------------------------------------------------------
extern "C" void kernel_launch(void* const* d_in, const int* in_sizes, int n_in,
                              void* d_out, int out_size, void* d_ws, size_t ws_size,
                              hipStream_t stream) {
    const float* h    = (const float*)d_in[0];
    const float* pos  = (const float*)d_in[1];
    const int*   ei   = (const int*)d_in[2];
    const int*   bat  = (const int*)d_in[3];
    const float* few  = (const float*)d_in[4];
    const float* feb  = (const float*)d_in[5];
    const int*   src  = ei;
    const int*   dst  = ei + NEDGE;

    char* ws = (char*)d_ws;
    size_t off = 0;
    auto alloc = [&](size_t bytes) -> char* {
        off = (off + 255) & ~(size_t)255;
        char* p = ws + off;
        off += bytes;
        return p;
    };
    float*    x    = (float*)alloc((size_t)NNODE * 64 * 4);
    _Float16* xh   = (_Float16*)alloc((size_t)NNODE * 64 * 2);
    float*    q    = (float*)alloc((size_t)NNODE * 3 * 4);
    float*    agg  = (float*)alloc((size_t)NNODE * 64 * 4);
    float*    tbuf = (float*)alloc((size_t)NNODE * 64 * 4);
    unsigned* wp1a = (unsigned*)alloc(12 * 32 * 8 * 4);
    unsigned* wp2a = (unsigned*)alloc(8 * 32 * 8 * 4);
    unsigned* wp1b = (unsigned*)alloc(12 * 32 * 8 * 4);
    unsigned* wp2b = (unsigned*)alloc(8 * 32 * 8 * 4);
    float*    stats = (float*)alloc(128 * 4);
    float*    pool  = (float*)alloc(NGRAPH * 64 * 4);
    float*    cnt   = (float*)alloc(NGRAPH * 4);

    const int nb = (NNODE + 255) / 256;
    const int ntiles = NEDGE / 16;

    // weight repack for both conv layers
    k_prepw<<<20, 256, 0, stream>>>((const float*)d_in[10], (const float*)d_in[12], wp1a, wp2a);
    k_prepw<<<20, 256, 0, stream>>>((const float*)d_in[24], (const float*)d_in[26], wp1b, wp2b);

    // embed
    k_embed<<<nb, 256, 0, stream>>>(h, few, feb, x, xh, NNODE);

    for (int layer = 0; layer < 2; ++layer) {
        int base = 6 + layer * 14;
        const float* hw1 = (const float*)d_in[base + 0];
        const float* hb1 = (const float*)d_in[base + 1];
        const float* hw2 = (const float*)d_in[base + 2];
        const float* hb2 = (const float*)d_in[base + 3];
        const float* fb1 = (const float*)d_in[base + 5];
        const float* fb2 = (const float*)d_in[base + 7];
        const float* gw1 = (const float*)d_in[base + 8];
        const float* gb1 = (const float*)d_in[base + 9];
        const float* ggm = (const float*)d_in[base + 10];
        const float* gbt = (const float*)d_in[base + 11];
        const float* gw2 = (const float*)d_in[base + 12];
        const float* gb2 = (const float*)d_in[base + 13];
        const unsigned* wp1 = layer ? wp1b : wp1a;
        const unsigned* wp2 = layer ? wp2b : wp2a;

        k_delta<<<nb, 256, 0, stream>>>(x, pos, hw1, hb1, hw2, hb2, q, NNODE);
        k_fill_u32<<<(NNODE * 64 + 255) / 256, 256, 0, stream>>>(
            (unsigned*)agg, 0xFF800000u, (long)NNODE * 64);
        k_edge<<<640, 256, 0, stream>>>(src, dst, pos, q, xh, wp1, wp2, fb1, fb2, agg, ntiles);
        k_fill_u32<<<1, 128, 0, stream>>>((unsigned*)stats, 0u, 128);
        k_g1<<<nb, 256, 0, stream>>>(agg, gw1, gb1, tbuf, stats, NNODE);
        k_bn<<<nb, 256, 0, stream>>>(tbuf, stats, ggm, gbt, gw2, gb2, x, xh, NNODE);
    }

    // pooling + readout
    k_fill_u32<<<(NGRAPH * 65 + 255) / 256, 256, 0, stream>>>((unsigned*)pool, 0u, NGRAPH * 64);
    k_fill_u32<<<1, 64, 0, stream>>>((unsigned*)cnt, 0u, NGRAPH);
    k_pool<<<nb, 256, 0, stream>>>(x, bat, pool, cnt, NNODE);
    k_readout<<<1, 64, 0, stream>>>(pool, cnt, (const float*)d_in[34], (const float*)d_in[35],
                                    (const float*)d_in[36], (const float*)d_in[37],
                                    (float*)d_out);
}
